// RelPosBlock_91250875171486
// MI455X (gfx1250) — compile-verified, matrix operand order checked by reading, not run on hardware
//
#include <hip/hip_runtime.h>
#include <math.h>

typedef __attribute__((ext_vector_type(16))) __bf16 v16bf;
typedef __attribute__((ext_vector_type(8)))  float  v8f;

union FragU { uint4 u[2]; v16bf v; };

__device__ __forceinline__ unsigned short f2bf(float f) {
  unsigned u = __float_as_uint(f);
  u += 0x7FFFu + ((u >> 16) & 1u);          // round-to-nearest-even
  return (unsigned short)(u >> 16);
}

__device__ __forceinline__ v8f z8() {
  v8f z;
#pragma unroll
  for (int i = 0; i < 8; i++) z[i] = 0.0f;
  return z;
}

__device__ __forceinline__ v8f wmma_bf16(v16bf a, v16bf b, v8f c) {
  return __builtin_amdgcn_wmma_f32_16x16x32_bf16(false, a, false, b, (short)0, c,
                                                 false, false);
}

// 16x32 bf16 WMMA fragment (A layout; B for x@w.T is identical since both are
// row-major with K contiguous).  Two 16B loads per lane.
__device__ __forceinline__ v16bf load_frag(const unsigned short* __restrict__ p,
                                           int ld, int rowBase, int kb, int lane) {
  int hi = lane >> 4;
  int r  = rowBase + (lane & 15);
  const unsigned short* base = p + (long)r * ld + kb + hi * 8;
  FragU f;
  f.u[0] = *(const uint4*)(base);           // k = hi*8 .. hi*8+7
  f.u[1] = *(const uint4*)(base + 16);      // k = 16+hi*8 .. 16+hi*8+7
  return f.v;
}

// ---------------------------------------------------------------- fp32 -> bf16
__global__ __launch_bounds__(256) void cvt_bf16_kernel(
    const float* __restrict__ in, unsigned short* __restrict__ out, long n) {
  long i = (long)blockIdx.x * 256 + threadIdx.x;
  if (i < n) out[i] = f2bf(in[i]);
}

// ------------------------------------------------------------- row LayerNorm
__global__ __launch_bounds__(256) void ln_bf16_kernel(
    const float* __restrict__ x, const float* __restrict__ w,
    const float* __restrict__ b, unsigned short* __restrict__ out, int C) {
  __shared__ float rs[256], rs2[256];
  int tid = threadIdx.x;
  const float* xr = x + (long)blockIdx.x * C;
  float v[4], s = 0.f, s2 = 0.f;
#pragma unroll
  for (int i = 0; i < 4; i++) {
    v[i] = xr[tid + i * 256];
    s += v[i]; s2 += v[i] * v[i];
  }
  rs[tid] = s; rs2[tid] = s2; __syncthreads();
  for (int o = 128; o; o >>= 1) {
    if (tid < o) { rs[tid] += rs[tid + o]; rs2[tid] += rs2[tid + o]; }
    __syncthreads();
  }
  float mean = rs[0] / C;
  float var  = rs2[0] / C - mean * mean;
  float rstd = rsqrtf(var + 1e-5f);
  unsigned short* orow = out + (long)blockIdx.x * C;
#pragma unroll
  for (int i = 0; i < 4; i++) {
    int c = tid + i * 256;
    orow[c] = f2bf((v[i] - mean) * rstd * w[c] + b[c]);
  }
}

// ------------------------------------------------------------------ bf16 GEMM
// OUT[M,N] = A[M,K] @ B[N,K]^T.  128 thr = 4 waves; wave -> 32x64 tile
// (2 A-frags + 4 B-frags -> 8 WMMAs per K-step), software-pipelined.
// EPI 0: fp32.  EPI 1: bias+GELU -> bf16.  EPI 2: resid + ls*(acc+bias) -> fp32.
template <int EPI, int KT>
__global__ __launch_bounds__(128) void gemm_bf16_kernel(
    const unsigned short* __restrict__ A, const unsigned short* __restrict__ Bw,
    const float* __restrict__ bias, const float* __restrict__ ls,
    const float* __restrict__ resid, void* __restrict__ outp,
    int M, int N) {
  int lane = threadIdx.x & 31, wave = threadIdx.x >> 5;
  int rowBase = blockIdx.y * 128 + wave * 32;
  int colBase = blockIdx.x * 64;
  v8f acc[8];
#pragma unroll
  for (int t = 0; t < 8; t++) acc[t] = z8();

  // prologue: kb = 0
  v16bf a0 = load_frag(A, KT, rowBase,      0, lane);
  v16bf a1 = load_frag(A, KT, rowBase + 16, 0, lane);
  v16bf b0 = load_frag(Bw, KT, colBase,      0, lane);
  v16bf b1 = load_frag(Bw, KT, colBase + 16, 0, lane);
  v16bf b2 = load_frag(Bw, KT, colBase + 32, 0, lane);
  v16bf b3 = load_frag(Bw, KT, colBase + 48, 0, lane);

  for (int kb = 0; kb < KT - 32; kb += 32) {
    int kn = kb + 32;
    // prefetch next K-step (distinct regs -> one clustered clause)
    v16bf na0 = load_frag(A, KT, rowBase,      kn, lane);
    v16bf na1 = load_frag(A, KT, rowBase + 16, kn, lane);
    v16bf nb0 = load_frag(Bw, KT, colBase,      kn, lane);
    v16bf nb1 = load_frag(Bw, KT, colBase + 16, kn, lane);
    v16bf nb2 = load_frag(Bw, KT, colBase + 32, kn, lane);
    v16bf nb3 = load_frag(Bw, KT, colBase + 48, kn, lane);

    acc[0] = wmma_bf16(a0, b0, acc[0]);
    acc[1] = wmma_bf16(a0, b1, acc[1]);
    acc[2] = wmma_bf16(a0, b2, acc[2]);
    acc[3] = wmma_bf16(a0, b3, acc[3]);
    acc[4] = wmma_bf16(a1, b0, acc[4]);
    acc[5] = wmma_bf16(a1, b1, acc[5]);
    acc[6] = wmma_bf16(a1, b2, acc[6]);
    acc[7] = wmma_bf16(a1, b3, acc[7]);

    a0 = na0; a1 = na1; b0 = nb0; b1 = nb1; b2 = nb2; b3 = nb3;
  }
  // tail K-step
  acc[0] = wmma_bf16(a0, b0, acc[0]);
  acc[1] = wmma_bf16(a0, b1, acc[1]);
  acc[2] = wmma_bf16(a0, b2, acc[2]);
  acc[3] = wmma_bf16(a0, b3, acc[3]);
  acc[4] = wmma_bf16(a1, b0, acc[4]);
  acc[5] = wmma_bf16(a1, b1, acc[5]);
  acc[6] = wmma_bf16(a1, b2, acc[6]);
  acc[7] = wmma_bf16(a1, b3, acc[7]);

  int hi = lane >> 4, ln = lane & 15;
  float bia[4], lsv[4];
#pragma unroll
  for (int t = 0; t < 4; t++) {
    int col = colBase + t * 16 + ln;
    bia[t] = (EPI != 0) ? bias[col] : 0.0f;
    lsv[t] = (EPI == 2) ? ls[col] : 0.0f;
  }
#pragma unroll
  for (int rt = 0; rt < 2; rt++) {
#pragma unroll
    for (int t = 0; t < 4; t++) {
      int col = colBase + t * 16 + ln;
#pragma unroll
      for (int r = 0; r < 8; r++) {
        int row = rowBase + rt * 16 + r + hi * 8;
        long idx = (long)row * N + col;
        float v = acc[rt * 4 + t][r];
        if (EPI == 0) {
          ((float*)outp)[idx] = v;
        } else if (EPI == 1) {
          float g = v + bia[t];
          g = 0.5f * g * (1.0f + erff(g * 0.70710678118654752f));
          ((unsigned short*)outp)[idx] = f2bf(g);
        } else {
          ((float*)outp)[idx] = resid[idx] + lsv[t] * (v + bia[t]);
        }
      }
    }
  }
}

// -------------------------------------------- QKV post: head LN, scale, V^T
__global__ __launch_bounds__(64) void qkv_post_kernel(
    const float* __restrict__ qkv, const float* __restrict__ qw,
    const float* __restrict__ qb, const float* __restrict__ kw,
    const float* __restrict__ kbv, unsigned short* __restrict__ qo,
    unsigned short* __restrict__ ko, unsigned short* __restrict__ vT,
    int N, int H, int D, int C) {
  __shared__ float sm[64], sm2[64];
  int h  = blockIdx.x % H;
  long bn = blockIdx.x / H;
  int  n  = (int)(bn % N);
  long b  = bn / N;
  int  d  = threadIdx.x;
  const float* base = qkv + bn * 3 * C;
  long bh = b * H + h;

  {
    float v = base[(0 * H + h) * D + d];
    sm[d] = v; sm2[d] = v * v; __syncthreads();
    for (int o = 32; o; o >>= 1) {
      if (d < o) { sm[d] += sm[d + o]; sm2[d] += sm2[d + o]; }
      __syncthreads();
    }
    float mean = sm[0] / D, var = sm2[0] / D - mean * mean;
    float r = rsqrtf(var + 1e-5f);
    qo[(bh * N + n) * D + d] = f2bf(((v - mean) * r * qw[d] + qb[d]) * 0.125f);
    __syncthreads();
  }
  {
    float v = base[(1 * H + h) * D + d];
    sm[d] = v; sm2[d] = v * v; __syncthreads();
    for (int o = 32; o; o >>= 1) {
      if (d < o) { sm[d] += sm[d + o]; sm2[d] += sm2[d + o]; }
      __syncthreads();
    }
    float mean = sm[0] / D, var = sm2[0] / D - mean * mean;
    float r = rsqrtf(var + 1e-5f);
    ko[(bh * N + n) * D + d] = f2bf((v - mean) * r * kw[d] + kbv[d]);
  }
  {
    float v = base[(2 * H + h) * D + d];
    vT[(bh * D + d) * N + n] = f2bf(v);
  }
}

// ----------------------------------------------------- flash attention + bias
// grid = B*H*(NN/64), 128 thr = 4 waves; wave -> 16 query rows.
template <int NN, int DD>
__global__ __launch_bounds__(128) void attn_kernel(
    const unsigned short* __restrict__ Q, const unsigned short* __restrict__ Km,
    const unsigned short* __restrict__ VT, const float* __restrict__ bias,
    unsigned short* __restrict__ O, int H, int C) {
  __shared__ __align__(16) unsigned short pbuf[4 * 16 * 32];
  const int nt = NN / 64;
  int bh   = blockIdx.x / nt;
  int qt   = blockIdx.x % nt;
  int h    = bh % H;
  int b    = bh / H;
  int lane = threadIdx.x & 31, wave = threadIdx.x >> 5;
  int hi   = lane >> 4, ln = lane & 15;
  int qBase = qt * 64 + wave * 16;

  const unsigned short* Qp = Q  + (long)bh * NN * DD;
  const unsigned short* Kp = Km + (long)bh * NN * DD;
  const unsigned short* Vp = VT + (long)bh * DD * NN;
  const float*          bp = bias + (long)h * NN * NN;

  v16bf q0 = load_frag(Qp, DD, qBase, 0, lane);
  v16bf q1 = load_frag(Qp, DD, qBase, 32, lane);

  v8f o[4];
#pragma unroll
  for (int t = 0; t < 4; t++) o[t] = z8();
  float mrun[8], lrun[8];
#pragma unroll
  for (int r = 0; r < 8; r++) { mrun[r] = -1e30f; lrun[r] = 0.0f; }

  unsigned short* pb = pbuf + wave * 16 * 32;

  for (int kt = 0; kt < NN; kt += 32) {
    // batched loads: 4 K-frags, 4 V-frags, 16 bias values (one clause each,
    // V + bias latency hides under the softmax VALU work)
    v16bf k00 = load_frag(Kp, DD, kt,       0, lane);
    v16bf k01 = load_frag(Kp, DD, kt,      32, lane);
    v16bf k10 = load_frag(Kp, DD, kt + 16,  0, lane);
    v16bf k11 = load_frag(Kp, DD, kt + 16, 32, lane);
    v16bf vf0 = load_frag(Vp, NN, 0,  kt, lane);
    v16bf vf1 = load_frag(Vp, NN, 16, kt, lane);
    v16bf vf2 = load_frag(Vp, NN, 32, kt, lane);
    v16bf vf3 = load_frag(Vp, NN, 48, kt, lane);
    float ba[16];
    int key0 = kt + ln, key1 = kt + 16 + ln;
#pragma unroll
    for (int r = 0; r < 8; r++) {
      int qr = qBase + r + hi * 8;
      ba[r]     = bp[(long)qr * NN + key0];
      ba[8 + r] = bp[(long)qr * NN + key1];
    }

    v8f s0 = z8(), s1 = z8();
    s0 = wmma_bf16(q0, k00, s0);
    s0 = wmma_bf16(q1, k01, s0);
    s1 = wmma_bf16(q0, k10, s1);
    s1 = wmma_bf16(q1, k11, s1);

    float sv[16];
#pragma unroll
    for (int r = 0; r < 8; r++) {
      sv[r]     = s0[r] + ba[r];
      sv[8 + r] = s1[r] + ba[8 + r];
    }
    // online softmax: row reductions across 16 lanes of each half-wave
    float mx[8];
#pragma unroll
    for (int r = 0; r < 8; r++) mx[r] = fmaxf(sv[r], sv[8 + r]);
    for (int off = 8; off; off >>= 1)
#pragma unroll
      for (int r = 0; r < 8; r++) mx[r] = fmaxf(mx[r], __shfl_xor(mx[r], off, 16));

    float f[8], rsum[8];
#pragma unroll
    for (int r = 0; r < 8; r++) {
      float mnew = fmaxf(mrun[r], mx[r]);
      f[r] = __expf(mrun[r] - mnew);
      mrun[r] = mnew;
      sv[r]     = __expf(sv[r]     - mnew);
      sv[8 + r] = __expf(sv[8 + r] - mnew);
      rsum[r] = sv[r] + sv[8 + r];
    }
    for (int off = 8; off; off >>= 1)
#pragma unroll
      for (int r = 0; r < 8; r++) rsum[r] += __shfl_xor(rsum[r], off, 16);
#pragma unroll
    for (int r = 0; r < 8; r++) lrun[r] = lrun[r] * f[r] + rsum[r];
#pragma unroll
    for (int t = 0; t < 4; t++)
#pragma unroll
      for (int r = 0; r < 8; r++) o[t][r] *= f[r];

    // P (C-layout) -> LDS -> A-fragment layout.  DS ops are in-order per wave
    // and each wave has a private 1KB region, so a dscnt wait suffices.
#pragma unroll
    for (int r = 0; r < 8; r++) {
      int m = r + hi * 8;
      pb[m * 32 + ln]      = f2bf(sv[r]);
      pb[m * 32 + 16 + ln] = f2bf(sv[8 + r]);
    }
    asm volatile("s_wait_dscnt 0" ::: "memory");
    FragU pf;
    {
      const unsigned short* base = pb + ln * 32 + hi * 8;
      pf.u[0] = *(const uint4*)(base);
      pf.u[1] = *(const uint4*)(base + 16);
    }
    asm volatile("" ::: "memory");

    o[0] = wmma_bf16(pf.v, vf0, o[0]);
    o[1] = wmma_bf16(pf.v, vf1, o[1]);
    o[2] = wmma_bf16(pf.v, vf2, o[2]);
    o[3] = wmma_bf16(pf.v, vf3, o[3]);
  }

  float inv[8];
#pragma unroll
  for (int r = 0; r < 8; r++) inv[r] = 1.0f / lrun[r];
#pragma unroll
  for (int t = 0; t < 4; t++) {
#pragma unroll
    for (int r = 0; r < 8; r++) {
      int qr  = qBase + r + hi * 8;
      int col = h * DD + t * 16 + ln;
      O[((long)(b * NN + qr)) * C + col] = f2bf(o[t][r] * inv[r]);
    }
  }
}

// ============================================================================
extern "C" void kernel_launch(void* const* d_in, const int* in_sizes, int n_in,
                              void* d_out, int out_size, void* d_ws, size_t ws_size,
                              hipStream_t stream) {
  (void)in_sizes; (void)n_in; (void)out_size; (void)ws_size;
  const float* x      = (const float*)d_in[0];
  const float* rbias  = (const float*)d_in[1];
  const float* qkv_w  = (const float*)d_in[2];
  const float* qnw    = (const float*)d_in[3];
  const float* qnb    = (const float*)d_in[4];
  const float* knw    = (const float*)d_in[5];
  const float* knb    = (const float*)d_in[6];
  const float* proj_w = (const float*)d_in[7];
  const float* proj_b = (const float*)d_in[8];
  const float* n1w    = (const float*)d_in[9];
  const float* n1b    = (const float*)d_in[10];
  const float* n2w    = (const float*)d_in[11];
  const float* n2b    = (const float*)d_in[12];
  const float* fc1_w  = (const float*)d_in[13];
  const float* fc1_b  = (const float*)d_in[14];
  const float* fc2_w  = (const float*)d_in[15];
  const float* fc2_b  = (const float*)d_in[16];
  const float* ls1    = (const float*)d_in[17];
  const float* ls2    = (const float*)d_in[18];

  const int B = 4, N = 1024, C = 1024, H = 16, D = 64, Hid = 4096;
  const int M = B * N;

  char* ws = (char*)d_ws;
  size_t off = 0;
  auto alloc = [&](size_t bytes) -> void* {
    void* p = ws + off;
    off += (bytes + 255) & ~(size_t)255;
    return p;
  };
  unsigned short* h_bf   = (unsigned short*)alloc((size_t)M * C * 2);
  unsigned short* w_qkv  = (unsigned short*)alloc((size_t)3 * C * C * 2);
  unsigned short* w_proj = (unsigned short*)alloc((size_t)C * C * 2);
  unsigned short* w_fc1  = (unsigned short*)alloc((size_t)Hid * C * 2);
  unsigned short* w_fc2  = (unsigned short*)alloc((size_t)C * Hid * 2);
  float*          qkv_f  = (float*)alloc((size_t)M * 3 * C * 4);
  unsigned short* q_bf   = (unsigned short*)alloc((size_t)M * C * 2);
  unsigned short* k_bf   = (unsigned short*)alloc((size_t)M * C * 2);
  unsigned short* vT_bf  = (unsigned short*)alloc((size_t)M * C * 2);
  unsigned short* o_bf   = (unsigned short*)alloc((size_t)M * C * 2);
  float*          x1     = (float*)alloc((size_t)M * C * 4);
  unsigned short* h2_bf  = (unsigned short*)alloc((size_t)M * C * 2);
  unsigned short* m1_bf  = (unsigned short*)alloc((size_t)M * Hid * 2);

  {
    long c1 = (long)3 * C * C, c2 = (long)C * C, c3 = (long)Hid * C, c4 = (long)C * Hid;
    cvt_bf16_kernel<<<(unsigned)((c1 + 255) / 256), 256, 0, stream>>>(qkv_w, w_qkv, c1);
    cvt_bf16_kernel<<<(unsigned)((c2 + 255) / 256), 256, 0, stream>>>(proj_w, w_proj, c2);
    cvt_bf16_kernel<<<(unsigned)((c3 + 255) / 256), 256, 0, stream>>>(fc1_w, w_fc1, c3);
    cvt_bf16_kernel<<<(unsigned)((c4 + 255) / 256), 256, 0, stream>>>(fc2_w, w_fc2, c4);
  }

  // LN1 -> h (bf16)
  ln_bf16_kernel<<<M, 256, 0, stream>>>(x, n1w, n1b, h_bf, C);

  // qkv = h @ qkv_w^T   (fp32 out)
  gemm_bf16_kernel<0, 1024><<<dim3(3 * C / 64, M / 128), 128, 0, stream>>>(
      h_bf, w_qkv, nullptr, nullptr, nullptr, qkv_f, M, 3 * C);

  // per-head QK LayerNorm, q*scale, V transpose
  qkv_post_kernel<<<B * N * H, 64, 0, stream>>>(qkv_f, qnw, qnb, knw, knb,
                                                q_bf, k_bf, vT_bf, N, H, D, C);

  // flash attention with rel-pos bias -> o (bf16, [B,N,C])
  attn_kernel<1024, 64><<<B * H * (N / 64), 128, 0, stream>>>(
      q_bf, k_bf, vT_bf, rbias, o_bf, H, C);

  // x1 = x + ls1 * (o @ proj_w^T + proj_b)
  gemm_bf16_kernel<2, 1024><<<dim3(C / 64, M / 128), 128, 0, stream>>>(
      o_bf, w_proj, proj_b, ls1, x, x1, M, C);

  // LN2 -> h2 (bf16)
  ln_bf16_kernel<<<M, 256, 0, stream>>>(x1, n2w, n2b, h2_bf, C);

  // m1 = gelu(h2 @ fc1_w^T + fc1_b)   (bf16 out)
  gemm_bf16_kernel<1, 1024><<<dim3(Hid / 64, M / 128), 128, 0, stream>>>(
      h2_bf, w_fc1, fc1_b, nullptr, nullptr, m1_bf, M, Hid);

  // out = x1 + ls2 * (m1 @ fc2_w^T + fc2_b)
  gemm_bf16_kernel<2, 4096><<<dim3(C / 64, M / 128), 128, 0, stream>>>(
      m1_bf, w_fc2, fc2_b, ls2, x1, (float*)d_out, M, C);
}
